// MirrorMemoryBank_62723702391273
// MI455X (gfx1250) — matrix operand order repaired and checked by reading, not compiled, not run on hardware
//
#include <hip/hip_runtime.h>
#include <hip/hip_bf16.h>
#include <math.h>
#include <stdint.h>

// ---------------------------------------------------------------------------
// MirrorMemoryBank on MI455X (gfx1250, wave32, WMMA)
//
//   1. qws = x @ Wq^T            (LDS-blocked WMMA f16 GEMM, f32 accum)
//   2. kws = memory @ Wk^T
//   3. vws = memory @ Wv^T
//   4. per (b,h): logits = scale * Q_h @ K_h^T -> sparse_attn region of d_out
//   5. fused top-k(128) + masked softmax + sparse @ V
//      (async GLOBAL_LOAD_ASYNC_TO_LDS row staging, radix select, ballot ties)
//   6. recon @ Wo^T + bo -> d_out[0 : B*N*C)
// ---------------------------------------------------------------------------

typedef __attribute__((ext_vector_type(16))) _Float16 v16h;
typedef __attribute__((ext_vector_type(8)))  _Float16 v8h;
typedef __attribute__((ext_vector_type(8)))  float    v8f;

#define Bb 2
#define Nn 2048
#define Cc 1024
#define Mm 4096
#define Hh 8
#define KK 128
#define HDd 128

// ---------------- LDS-blocked WMMA GEMM ------------------------------------
// Y[M,N] = scale * (A[M,K] @ B[N,K]^T) + bias[N]
// Block tile 64(M) x 128(N), K-step 64 (2 WMMA K-chunks per stage).
// 8 waves: 4 along M x 2 along N; each wave computes 16x64 via 4 accumulators.
// Per barrier pair: 8 v_wmma_f32_16x16x32_f16 (A fragment reused 4x per chunk).
#define BM 64
#define BN 128
#define BK 64
#define LDSTR 72   // padded row stride in halves (36 banks/row, conflict-free)

__global__ __launch_bounds__(256) void gemm_nt_wmma(
    const float* __restrict__ A, int lda,
    const float* __restrict__ Bm, int ldb,
    float* __restrict__ Y, int ldy,
    int Mrows, int Ncols, int Kdim,
    float scale, const float* __restrict__ bias) {
  __shared__ _Float16 As[BM * LDSTR];   //  9.2 KB
  __shared__ _Float16 Bs[BN * LDSTR];   // 18.4 KB

  int tid = threadIdx.x;
  int lane = tid & 31;
  int wave = tid >> 5;
  int wm = wave & 3;        // wave M index (0..3)
  int wn = wave >> 2;       // wave N index (0..1)
  int nblk = Ncols / BN;
  int bm0 = (blockIdx.x / nblk) * BM;
  int bn0 = (blockIdx.x % nblk) * BN;

  v8f acc[4] = {};

  for (int k0 = 0; k0 < Kdim; k0 += BK) {
    // ---- batch ALL staging loads first (one load clause, no per-load waits)
    // A: 64x64 f32 = 1024 float4, 4/thread. B: 128x64 = 2048 float4, 8/thread.
    float4 fa[4];
#pragma unroll
    for (int t = 0; t < 4; ++t) {
      int idx = tid + t * 256;
      int r = idx >> 4;                 // 16 float4 per 64-wide row
      int c4 = (idx & 15) << 2;
      const float* src = A + (size_t)(bm0 + r) * lda + k0 + c4;
      if (k0 + BK < Kdim) __builtin_prefetch(src + BK, 0, 1);  // global_prefetch_b8
      fa[t] = *(const float4*)src;
    }
    float4 fb[8];
#pragma unroll
    for (int t = 0; t < 8; ++t) {
      int idx = tid + t * 256;
      int r = idx >> 4;
      int c4 = (idx & 15) << 2;
      const float* src = Bm + (size_t)(bn0 + r) * ldb + k0 + c4;
      if (k0 + BK < Kdim) __builtin_prefetch(src + BK, 0, 1);
      fb[t] = *(const float4*)src;
    }
    // ---- convert f32 -> f16 and store to LDS
#pragma unroll
    for (int t = 0; t < 4; ++t) {
      int idx = tid + t * 256;
      int r = idx >> 4;
      int c4 = (idx & 15) << 2;
      _Float16* d = &As[r * LDSTR + c4];
      d[0] = (_Float16)fa[t].x; d[1] = (_Float16)fa[t].y;
      d[2] = (_Float16)fa[t].z; d[3] = (_Float16)fa[t].w;
    }
#pragma unroll
    for (int t = 0; t < 8; ++t) {
      int idx = tid + t * 256;
      int r = idx >> 4;
      int c4 = (idx & 15) << 2;
      _Float16* d = &Bs[r * LDSTR + c4];
      d[0] = (_Float16)fb[t].x; d[1] = (_Float16)fb[t].y;
      d[2] = (_Float16)fb[t].z; d[3] = (_Float16)fb[t].w;
    }
    __syncthreads();

    // ---- two 32-deep WMMA chunks per stage
    int ar = wm * 16 + (lane & 15);
    int kb = (lane < 16) ? 0 : 8;       // A frag: chunks [kb,kb+8) & [16+kb,+8)
    int kb2 = (lane < 16) ? 0 : 8;      // B frag: chunks [kb2,kb2+8) & [16+kb2,+8)
#pragma unroll
    for (int kc = 0; kc < 2; ++kc) {
      int ko = kc * 32;
      v8h alo = *(const v8h*)&As[ar * LDSTR + ko + kb];
      v8h ahi = *(const v8h*)&As[ar * LDSTR + ko + 16 + kb];
      v16h afrag = __builtin_shufflevector(alo, ahi,
          0, 1, 2, 3, 4, 5, 6, 7, 8, 9, 10, 11, 12, 13, 14, 15);
#pragma unroll
      for (int t = 0; t < 4; ++t) {
        int br = wn * 64 + t * 16 + (lane & 15);
        v8h blo = *(const v8h*)&Bs[br * LDSTR + ko + kb2];
        v8h bhi = *(const v8h*)&Bs[br * LDSTR + ko + 16 + kb2];
        v16h bfrag = __builtin_shufflevector(blo, bhi,
            0, 1, 2, 3, 4, 5, 6, 7, 8, 9, 10, 11, 12, 13, 14, 15);
        acc[t] = __builtin_amdgcn_wmma_f32_16x16x32_f16(
            false, afrag, false, bfrag, (short)0, acc[t], false, false);
      }
    }
    __syncthreads();
  }

  // ---- store: 32-bit C/D layout: lane -> N=lane&15, VGPR i -> M=i+(lane<16?0:8)
  int mb = (lane < 16) ? 0 : 8;
  int mrow0 = bm0 + wm * 16 + mb;
#pragma unroll
  for (int t = 0; t < 4; ++t) {
    int nc = bn0 + wn * 64 + t * 16 + (lane & 15);
    float bv = bias ? bias[nc] : 0.0f;
#pragma unroll
    for (int i = 0; i < 8; ++i)
      Y[(size_t)(mrow0 + i) * ldy + nc] = acc[t][i] * scale + bv;
  }
}

// ---------------- fused top-k / softmax / sparse-AV ------------------------
__device__ __forceinline__ unsigned mono_key(float f) {
  unsigned u = __float_as_uint(f);
  return (u & 0x80000000u) ? ~u : (u | 0x80000000u);  // order-preserving
}

__device__ __forceinline__ unsigned ballot32(bool p) {
#if __has_builtin(__builtin_amdgcn_ballot_w32)
  return __builtin_amdgcn_ballot_w32(p);
#else
  return (unsigned)__ballot(p);
#endif
}

#define ROWS_PER_BLK 4
#define OFF_HIST  (Mm * 4)            // 16384 B of row data
#define OFF_SIDX  (OFF_HIST + 1024)   // 256 x u32 histogram
#define OFF_SELP  (OFF_SIDX + 1024)   // 256 x i32 selected indices
#define OFF_SC    (OFF_SELP + 1024)   // 256 x f32 selected probabilities
#define ROWBYTES  (OFF_SC + 32)       // + broadcast scalars

__global__ __launch_bounds__(128) void topk_softmax_av(
    float* __restrict__ attn,            // [B*H*N, M] logits in, probs out
    const float* __restrict__ vws,       // [M, C]
    float* __restrict__ recon) {         // [B*N, C]
  extern __shared__ unsigned char smem_raw[];
  int lane = threadIdx.x & 31;
  int w = threadIdx.x >> 5;
  unsigned char* base = smem_raw + (size_t)w * ROWBYTES;
  float*    rowv   = (float*)base;
  unsigned* hist   = (unsigned*)(base + OFF_HIST);
  int*      selIdx = (int*)(base + OFF_SIDX);
  float*    selP   = (float*)(base + OFF_SELP);
  int*      sc     = (int*)(base + OFF_SC);   // [0]=digit [1]=kRemain [2]=selCount

  int row = blockIdx.x * ROWS_PER_BLK + w;    // row = (b*H + h)*N + n
  int n  = row & (Nn - 1);
  int bh = row >> 11;
  int h  = bh & (Hh - 1);
  int b  = bh >> 3;
  float* Lrow = attn + (size_t)row * Mm;

  // ---- stage row HBM -> LDS via CDNA5 async DMA (ASYNCcnt path).
  // 16B per lane per issue; 32 issues cover 4096 floats. No VGPR data.
  for (int i = lane * 4; i < Mm; i += 128) {
    unsigned lds_off = (unsigned)(uintptr_t)(void*)(rowv + i);
    unsigned long long ga = (unsigned long long)(uintptr_t)(const void*)(Lrow + i);
    asm volatile("global_load_async_to_lds_b128 %0, %1, off"
                 :: "v"(lds_off), "v"(ga) : "memory");
  }
  asm volatile("s_wait_asynccnt 0x0" ::: "memory");

  // row max
  float mx = -3.4e38f;
  for (int i = lane; i < Mm; i += 32) mx = fmaxf(mx, rowv[i]);
#pragma unroll
  for (int off = 16; off > 0; off >>= 1)
    mx = fmaxf(mx, __shfl_xor(mx, off, 32));
  if (lane == 0) sc[2] = 0;
  __syncthreads();

  // ---- exact K-th largest via 4-pass radix select on monotone keys
  unsigned prefixVal = 0, prefixMask = 0;
  int kRemain = KK;
#pragma unroll 1
  for (int pass = 0; pass < 4; ++pass) {
    int shift = 24 - pass * 8;
    for (int i = lane; i < 256; i += 32) hist[i] = 0;
    __syncthreads();
    for (int i = lane; i < Mm; i += 32) {
      unsigned key = mono_key(rowv[i]);
      if ((key & prefixMask) == prefixVal)
        atomicAdd(&hist[(key >> shift) & 255u], 1u);
    }
    __syncthreads();
    if (lane == 0) {
      int accn = 0, d = 255;
      for (; d > 0; --d) {
        int cdx = (int)hist[d];
        if (accn + cdx >= kRemain) break;
        accn += cdx;
      }
      sc[0] = d;
      sc[1] = kRemain - accn;
    }
    __syncthreads();
    prefixVal |= ((unsigned)sc[0]) << shift;
    prefixMask |= 0xFFu << shift;
    kRemain = sc[1];
  }
  unsigned Tkey = prefixVal;      // key of the K-th largest element
  int tiesNeeded = kRemain;       // #elems == Tkey to take, in index order

  // ---- pass A: softmax denominator over the selected set
  float sum = 0.0f;
  int tieBase = 0;
  for (int basei = 0; basei < Mm; basei += 32) {
    int i = basei + lane;
    float fv = rowv[i];
    unsigned key = mono_key(fv);
    bool eq = (key == Tkey);
    unsigned bal = ballot32(eq);
    int pre = __popc(bal & ((1u << lane) - 1u));
    bool sel = (key > Tkey) || (eq && (tieBase + pre) < tiesNeeded);
    tieBase += __popc(bal);
    if (sel) sum += __expf(fv - mx);
  }
#pragma unroll
  for (int off = 16; off > 0; off >>= 1) sum += __shfl_xor(sum, off, 32);
  float inv = 1.0f / sum;

  // ---- pass B: overwrite row with probabilities, collect (m, p) pairs
  tieBase = 0;
  for (int basei = 0; basei < Mm; basei += 32) {
    int i = basei + lane;
    float fv = rowv[i];
    unsigned key = mono_key(fv);
    bool eq = (key == Tkey);
    unsigned bal = ballot32(eq);
    int pre = __popc(bal & ((1u << lane) - 1u));
    bool sel = (key > Tkey) || (eq && (tieBase + pre) < tiesNeeded);
    tieBase += __popc(bal);
    float p = sel ? __expf(fv - mx) * inv : 0.0f;
    Lrow[i] = p;
    if (sel) {
      int pos = atomicAdd(&sc[2], 1);
      if (pos < 256) { selIdx[pos] = i; selP[pos] = p; }
    }
  }
  __syncthreads();

  // ---- sparse A@V: ~128 selected rows x 128 dims; 4 dims per lane
  int cnt = sc[2]; if (cnt > 256) cnt = 256;
  int d0 = lane * 4;
  float ax = 0.f, ay = 0.f, az = 0.f, aw = 0.f;
  const float* vbase = vws + (size_t)h * HDd + d0;
  for (int j = 0; j < cnt; ++j) {
    int m = selIdx[j];
    float p = selP[j];
    const float4 vv = *(const float4*)(vbase + (size_t)m * Cc);
    ax += p * vv.x; ay += p * vv.y; az += p * vv.z; aw += p * vv.w;
  }
  float4 o; o.x = ax; o.y = ay; o.z = az; o.w = aw;
  *(float4*)(recon + ((size_t)(b * Nn + n)) * Cc + h * HDd + d0) = o;
}

// ---------------------------------------------------------------------------
extern "C" void kernel_launch(void* const* d_in, const int* in_sizes, int n_in,
                              void* d_out, int out_size, void* d_ws, size_t ws_size,
                              hipStream_t stream) {
  (void)in_sizes; (void)n_in; (void)out_size; (void)ws_size;
  const float* x   = (const float*)d_in[0];
  const float* mem = (const float*)d_in[1];
  const float* Wq  = (const float*)d_in[2];
  const float* Wk  = (const float*)d_in[3];
  const float* Wv  = (const float*)d_in[4];
  const float* Wo  = (const float*)d_in[5];
  const float* bo  = (const float*)d_in[6];

  float* outp = (float*)d_out;
  float* attn = outp + (size_t)Bb * Nn * Cc;       // sparse_attn region (& logits scratch)

  float* ws  = (float*)d_ws;                        // 64 MB of f32 scratch
  float* qws = ws;
  float* kws = qws + (size_t)Bb * Nn * Cc;
  float* vws = kws + (size_t)Mm * Cc;
  float* rws = vws + (size_t)Mm * Cc;

  const float rscale = 0.08838834764831845f;        // 1/sqrt(HD)
  dim3 blk(256);

  // q/k/v projections: 64x128 block tiles, K-step 64
  int grid_qo = ((Bb * Nn) / BM) * (Cc / BN);       // 512
  gemm_nt_wmma<<<dim3(grid_qo), blk, 0, stream>>>(
      x, Cc, Wq, Cc, qws, Cc, Bb * Nn, Cc, Cc, 1.0f, (const float*)nullptr);
  int grid_kv = (Mm / BM) * (Cc / BN);              // 512
  gemm_nt_wmma<<<dim3(grid_kv), blk, 0, stream>>>(
      mem, Cc, Wk, Cc, kws, Cc, Mm, Cc, Cc, 1.0f, (const float*)nullptr);
  gemm_nt_wmma<<<dim3(grid_kv), blk, 0, stream>>>(
      mem, Cc, Wv, Cc, vws, Cc, Mm, Cc, Cc, 1.0f, (const float*)nullptr);

  // per-(b,h) logits GEMM into the sparse_attn region (in-place scratch)
  int grid_l = (Nn / BM) * (Mm / BN);               // 1024
  for (int b = 0; b < Bb; ++b)
    for (int h = 0; h < Hh; ++h) {
      const float* Aq = qws + (size_t)b * Nn * Cc + h * HDd;
      const float* Bk = kws + h * HDd;
      float* Lo = attn + ((size_t)(b * Hh + h)) * Nn * Mm;
      gemm_nt_wmma<<<dim3(grid_l), blk, 0, stream>>>(
          Aq, Cc, Bk, Cc, Lo, Mm, Nn, Mm, HDd, rscale, (const float*)nullptr);
    }

  // fused top-k / softmax / sparse AV (78 KB dynamic LDS: gfx1250 320KB/WGP)
  size_t smem = (size_t)ROWBYTES * ROWS_PER_BLK;
  (void)hipFuncSetAttribute((const void*)topk_softmax_av,
                            hipFuncAttributeMaxDynamicSharedMemorySize, (int)smem);
  topk_softmax_av<<<dim3((Bb * Hh * Nn) / ROWS_PER_BLK), dim3(128), smem, stream>>>(
      attn, vws, rws);

  // output projection with bias
  gemm_nt_wmma<<<dim3(grid_qo), blk, 0, stream>>>(
      rws, Cc, Wo, Cc, outp, Cc, Bb * Nn, Cc, Cc, 1.0f, bo);
}
